// NLP_Movie_TF_RNN_11269994185039
// MI455X (gfx1250) — compile-verified, matrix operand order checked by reading, not run on hardware
//
#include <hip/hip_runtime.h>
#include <hip/hip_bf16.h>
#include <math.h>

// Problem constants (from reference)
#define VOCAB 30000
#define EMB   100
#define HID   128
#define BATCH 1024
#define SEQ   512
#define TILE_B 16          // batch rows per workgroup (one WMMA M-tile)
#define KPAD  128          // EMB padded to a multiple of 32 for WMMA K
#define XPITCH (KPAD + 8)  // LDS row pitch (bf16) to avoid bank conflicts
#define HPITCH (HID + 8)

typedef __bf16 bf16_t;
typedef bf16_t v16bf __attribute__((ext_vector_type(16)));
typedef float  v8f   __attribute__((ext_vector_type(8)));

__device__ __forceinline__ float fast_tanh(float x) {
#if __has_builtin(__builtin_amdgcn_tanhf)
    return __builtin_amdgcn_tanhf(x);   // gfx1250 V_TANH_F32 (TRANS unit)
#else
    return tanhf(x);
#endif
}

__global__ __launch_bounds__(256, 1) void rnn_fused_kernel(
    const int*   __restrict__ X,   // [B, T] int32 token ids
    const float* __restrict__ E,   // [VOCAB, EMB]
    const float* __restrict__ Wx,  // [EMB, HID]
    const float* __restrict__ Wh,  // [HID, HID]
    const float* __restrict__ b,   // [HID]
    const float* __restrict__ Wd,  // [HID, 1]
    const float* __restrict__ bd,  // [1]
    float*       __restrict__ out) // [B, 1]
{
    __shared__ bf16_t wstage[HID * HID];              // 32 KB: weight staging (startup only)
    __shared__ bf16_t xbuf[2][TILE_B][XPITCH];        // double-buffered emb tile (bf16)
    __shared__ bf16_t hbuf[2][TILE_B][HPITCH];        // double-buffered hidden state (bf16)
    __shared__ float  hf32[TILE_B][HID + 1];          // final h in f32 for the output head

    const int tid  = threadIdx.x;
    const int lane = tid & 31;           // wave32
    const int wave = tid >> 5;           // 0..7; wave owns hidden cols [16w, 16w+16)
    const int half = (lane >> 4) & 1;    // lane half selects K sub-pattern / M offset
    const int nl   = lane & 15;
    const int n    = wave * 16 + nl;     // this lane's hidden output column
    const int b0   = blockIdx.x * TILE_B;

    // ---------------- Stage Wh -> bf16, build resident B-fragments ----------------
    for (int i = tid; i < HID * HID; i += 256) wstage[i] = (bf16_t)Wh[i];
    __syncthreads();
    v16bf whf[4];
#pragma unroll
    for (int c = 0; c < 4; ++c) {
#pragma unroll
        for (int j = 0; j < 8; ++j) {
            whf[c][j]     = wstage[(c * 32 +      half * 8 + j) * HID + n];
            whf[c][8 + j] = wstage[(c * 32 + 16 + half * 8 + j) * HID + n];
        }
    }
    __syncthreads();
    // ---------------- Stage Wx (zero-pad K from 100 to 128) ----------------
    for (int i = tid; i < HID * HID; i += 256) {
        int k = i >> 7;                  // row (K) index, pitch 128 == HID
        wstage[i] = (k < EMB) ? (bf16_t)Wx[i] : (bf16_t)0.0f;
    }
    __syncthreads();
    v16bf wxf[4];
#pragma unroll
    for (int c = 0; c < 4; ++c) {
#pragma unroll
        for (int j = 0; j < 8; ++j) {
            wxf[c][j]     = wstage[(c * 32 +      half * 8 + j) * HID + n];
            wxf[c][8 + j] = wstage[(c * 32 + 16 + half * 8 + j) * HID + n];
        }
    }

    // ---------------- Zero h0 and emb-buffer padding ----------------
    for (int i = tid; i < 2 * TILE_B * HPITCH; i += 256) (&hbuf[0][0][0])[i] = (bf16_t)0.0f;
    for (int i = tid; i < 2 * TILE_B * XPITCH; i += 256) (&xbuf[0][0][0])[i] = (bf16_t)0.0f;
    const float bias = b[n];

    // Gather mapping: 16 threads per batch row, strided over the 100 emb cols
    const int grow = tid >> 4;   // 0..15 : batch row within tile
    const int gcol = tid & 15;
    float pend[7];
    // Prefetch embeddings for t = 0
    {
        int tok = X[(b0 + grow) * SEQ + 0];
        const float* er = E + (long)tok * EMB;
#pragma unroll
        for (int i = 0; i < 7; ++i) {
            int cc = gcol + 16 * i;
            pend[i] = (cc < EMB) ? er[cc] : 0.0f;
        }
    }
    __syncthreads();

    // Commit the pipelined embedding registers for step t into xbuf[cur]
    auto commit_x = [&](int cur) {
#pragma unroll
        for (int i = 0; i < 7; ++i) {
            int cc = gcol + 16 * i;
            if (cc < EMB) xbuf[cur][grow][cc] = (bf16_t)pend[i];
        }
    };

    // Load A fragments + run the two independent 4-WMMA chains for one step.
    auto matstep = [&](int cur, v8f& accx, v8f& acch) {
        v16bf ax[4], ah[4];
#pragma unroll
        for (int c = 0; c < 4; ++c) {
#pragma unroll
            for (int j = 0; j < 8; ++j) {
                ax[c][j]     = xbuf[cur][nl][c * 32 +      half * 8 + j];
                ax[c][8 + j] = xbuf[cur][nl][c * 32 + 16 + half * 8 + j];
                ah[c][j]     = hbuf[cur][nl][c * 32 +      half * 8 + j];
                ah[c][8 + j] = hbuf[cur][nl][c * 32 + 16 + half * 8 + j];
            }
        }
#pragma unroll
        for (int e = 0; e < 8; ++e) { accx[e] = bias; acch[e] = 0.0f; }
#pragma unroll
        for (int c = 0; c < 4; ++c) {
            accx = __builtin_amdgcn_wmma_f32_16x16x32_bf16(
                false, ax[c], false, wxf[c], (short)0, accx, false, false);
            acch = __builtin_amdgcn_wmma_f32_16x16x32_bf16(
                false, ah[c], false, whf[c], (short)0, acch, false, false);
        }
    };

    // ---------------- Serial recurrence (steps 0 .. SEQ-2): branch-free body ----------------
    for (int t = 0; t < SEQ - 1; ++t) {
        const int cur = t & 1;
        commit_x(cur);                       // emb tile for step t
        {                                    // HBM gather for step t+1 (always in range)
            int tok = X[(b0 + grow) * SEQ + (t + 1)];
            const float* er = E + (long)tok * EMB;
#pragma unroll
            for (int i = 0; i < 7; ++i) {
                int cc = gcol + 16 * i;
                pend[i] = (cc < EMB) ? er[cc] : 0.0f;
            }
        }
        __syncthreads();   // xbuf[cur] committed; hbuf[cur] from step t-1 visible

        v8f accx, acch;
        matstep(cur, accx, acch);

        // h_{t+1} = tanh(accx + acch); C/D layout: VGPR e -> row e + 8*half, col n
#pragma unroll
        for (int e = 0; e < 8; ++e) {
            float hv = fast_tanh(accx[e] + acch[e]);
            hbuf[cur ^ 1][e + 8 * half][n] = (bf16_t)hv;
        }
    }

    // ---------------- Peeled final step (t = SEQ-1): capture h in f32 ----------------
    {
        const int cur = (SEQ - 1) & 1;
        commit_x(cur);
        __syncthreads();

        v8f accx, acch;
        matstep(cur, accx, acch);

#pragma unroll
        for (int e = 0; e < 8; ++e) {
            float hv = fast_tanh(accx[e] + acch[e]);
            hf32[e + 8 * half][n] = hv;
        }
    }
    __syncthreads();

    // ---------------- Output head: sigmoid(h @ Wd + bd), one thread per row ----------------
    if (tid < TILE_B) {
        float acc = bd[0];
#pragma unroll 8
        for (int k = 0; k < HID; ++k) acc += hf32[tid][k] * Wd[k];
        out[b0 + tid] = 1.0f / (1.0f + expf(-acc));
    }
}

extern "C" void kernel_launch(void* const* d_in, const int* in_sizes, int n_in,
                              void* d_out, int out_size, void* d_ws, size_t ws_size,
                              hipStream_t stream) {
    const int*   X  = (const int*)  d_in[0];
    const float* E  = (const float*)d_in[1];
    const float* Wx = (const float*)d_in[2];
    const float* Wh = (const float*)d_in[3];
    const float* b  = (const float*)d_in[4];
    const float* Wd = (const float*)d_in[5];
    const float* bd = (const float*)d_in[6];
    float* out = (float*)d_out;

    rnn_fused_kernel<<<BATCH / TILE_B, 256, 0, stream>>>(X, E, Wx, Wh, b, Wd, bd, out);
}
